// GraphEncoder_25323127177729
// MI455X (gfx1250) — matrix operand order repaired and checked by reading, not compiled, not run on hardware
//
#include <hip/hip_runtime.h>

typedef __attribute__((ext_vector_type(16))) __bf16 v16bf;
typedef __attribute__((ext_vector_type(8)))  __bf16 v8bf;
typedef __attribute__((ext_vector_type(8)))  float  v8f;

constexpr int kH = 128;    // hidden dim
constexpr int kE = 64;     // output embedding dim
constexpr float kLnEps = 1e-5f;

#define TILE_M 64

// Build a 16x32 bf16 A fragment from row pointer `ap` (row-major, stride K):
// lo-lanes hold K 0..7 / 16..23, hi-lanes hold K 8..15 / 24..31.
__device__ __forceinline__ v16bf afrag(const __bf16* __restrict__ ap, int k0, int hi) {
  v8bf lo = *(const v8bf*)(ap + k0 + hi * 8);
  v8bf h8 = *(const v8bf*)(ap + k0 + 16 + hi * 8);
  return __builtin_shufflevector(lo, h8, 0,1,2,3,4,5,6,7,8,9,10,11,12,13,14,15);
}

// C[M x N] = A[M x K] @ B[K x N] (+ A2 @ B2) + bias.
// A, A2: bf16 row-major [M x K]. BT, B2T: bf16 TRANSPOSED [N x K] (col-major weights).
// C: f32 out; Cbf: optional bf16 shadow of C.
// Launch: grid.x = ceil(M/64), block = 32 * (N/16). N in {64,128}, K%32==0, M%16==0.
__global__ void __launch_bounds__(256)
gemm_wmma_bf16(const __bf16* __restrict__ A,  const __bf16* __restrict__ BT,
               const __bf16* __restrict__ A2, const __bf16* __restrict__ B2T,
               const float* __restrict__ bias, float* __restrict__ C,
               __bf16* __restrict__ Cbf,
               int M, int N, int K, int K2) {
  const int lane = threadIdx.x & 31;
  const int wave = threadIdx.x >> 5;    // N-tile index
  const int r    = lane & 15;           // A row / B col within 16-tile
  const int hi   = lane >> 4;           // lane-half selector
  const int row0 = blockIdx.x * TILE_M;
  const int col  = wave * 16 + r;       // global C/B column for this lane

  // Clamp per-subtile rows (uniform selects: M%16==0 so whole tiles are in/out;
  // out-of-range tiles compute garbage that the store guards discard).
  int r0 = row0      + r; if (row0      >= M) r0 = r;
  int r1 = row0 + 16 + r; if (row0 + 16 >= M) r1 = r;
  int r2 = row0 + 32 + r; if (row0 + 32 >= M) r2 = r;
  int r3 = row0 + 48 + r; if (row0 + 48 >= M) r3 = r;

  const float bv = bias ? bias[col] : 0.0f;
  v8f acc0, acc1, acc2, acc3;
  #pragma unroll
  for (int v = 0; v < 8; ++v) { acc0[v] = bv; acc1[v] = bv; acc2[v] = bv; acc3[v] = bv; }

  const __bf16* Ap = A; const __bf16* Bp = BT; int Kc = K;
  for (int pass = 0; pass < 2; ++pass) {
    if (Ap != nullptr) {
      const __bf16* a0p = Ap + (size_t)r0 * Kc;
      const __bf16* a1p = Ap + (size_t)r1 * Kc;
      const __bf16* a2p = Ap + (size_t)r2 * Kc;
      const __bf16* a3p = Ap + (size_t)r3 * Kc;
      const __bf16* bp  = Bp + (size_t)col * Kc;   // transposed: contiguous K per column
      for (int k0 = 0; k0 < Kc; k0 += 32) {
        if (k0 + 32 < Kc) __builtin_prefetch(a0p + k0 + 32, 0, 1);
        // B fragment: 2 runs of 8 contiguous bf16 (same k order as A) -> 2x global_load_b128 (reused by 4 WMMAs)
        v16bf b  = afrag(bp, k0, hi);
        v16bf a0 = afrag(a0p, k0, hi);
        v16bf a1 = afrag(a1p, k0, hi);
        v16bf a2 = afrag(a2p, k0, hi);
        v16bf a3 = afrag(a3p, k0, hi);
        // 4 independent WMMAs: straight-line, EXEC all-ones, no D->A/B hazards
        acc0 = __builtin_amdgcn_wmma_f32_16x16x32_bf16(false, a0, false, b, (short)0, acc0, false, false);
        acc1 = __builtin_amdgcn_wmma_f32_16x16x32_bf16(false, a1, false, b, (short)0, acc1, false, false);
        acc2 = __builtin_amdgcn_wmma_f32_16x16x32_bf16(false, a2, false, b, (short)0, acc2, false, false);
        acc3 = __builtin_amdgcn_wmma_f32_16x16x32_bf16(false, a3, false, b, (short)0, acc3, false, false);
      }
    }
    Ap = A2; Bp = B2T; Kc = K2;
  }

  // C layout: VGPR v -> row = v + 8*hi (within 16-tile), col = lane&15.
  // Store guards are scalar-uniform (blockIdx-based).
  #pragma unroll
  for (int t = 0; t < 4; ++t) {
    const v8f& a = (t == 0) ? acc0 : (t == 1) ? acc1 : (t == 2) ? acc2 : acc3;
    int g = row0 + t * 16;
    if (g < M) {
      #pragma unroll
      for (int v = 0; v < 8; ++v) {
        size_t idx = (size_t)(g + v + hi * 8) * N + col;
        C[idx] = a[v];
        if (Cbf) Cbf[idx] = (__bf16)a[v];
      }
    }
  }
}

// One-time f32 -> bf16 conversion (inputs).
__global__ void f32_to_bf16(const float* __restrict__ x, __bf16* __restrict__ y, long long n) {
  long long i = (long long)blockIdx.x * blockDim.x + threadIdx.x;
  if (i < n) y[i] = (__bf16)x[i];
}

// Weight convert + transpose: W[K x N] f32 -> WT[N x K] bf16.
__global__ void transpose_w_bf16(const float* __restrict__ W, __bf16* __restrict__ WT,
                                 int K, int N) {
  int idx = blockIdx.x * blockDim.x + threadIdx.x;
  if (idx >= K * N) return;
  int k = idx / N, c = idx - k * N;            // coalesced read along columns
  WT[(size_t)c * K + k] = (__bf16)W[idx];
}

// One thread per (edge, feature). edge is [2 x n_edge]: row0 = src, row1 = dst.
__global__ void edge_aggregate(const float* __restrict__ h_src, const int* __restrict__ edge,
                               float* __restrict__ agg, float* __restrict__ cnt, int n_edge) {
  long long tid = (long long)blockIdx.x * blockDim.x + threadIdx.x;
  int e = (int)(tid >> 7);
  int f = (int)(tid & 127);
  if (e >= n_edge) return;
  int s = edge[e];
  int d = edge[n_edge + e];
  atomicAdd(&agg[(size_t)d * kH + f], h_src[(size_t)s * kH + f]);
  if (f == 0) atomicAdd(&cnt[d], 1.0f);
}

// mean = agg / max(cnt,1), emitted directly as bf16 (GEMM operand).
__global__ void mean_div(const float* __restrict__ agg, const float* __restrict__ cnt,
                         __bf16* __restrict__ mbf, int n) {
  long long tid = (long long)blockIdx.x * blockDim.x + threadIdx.x;
  if (tid >= (long long)n * kH) return;
  int node = (int)(tid >> 7);
  mbf[tid] = (__bf16)(agg[tid] / fmaxf(cnt[node], 1.0f));
}

// h[row] += LN(relu(xnew[row])) * gamma + beta; also writes bf16 shadow of h.
// One wave per 128-wide row.
__global__ void relu_ln_res(const float* __restrict__ xnew, float* __restrict__ h,
                            __bf16* __restrict__ hbf,
                            const float* __restrict__ gamma, const float* __restrict__ beta,
                            int n) {
  int row = blockIdx.x * (blockDim.x >> 5) + (threadIdx.x >> 5);
  if (row >= n) return;
  int lane = threadIdx.x & 31;
  float v[4];
  float s = 0.0f;
  #pragma unroll
  for (int j = 0; j < 4; ++j) {
    float x = xnew[(size_t)row * kH + lane + j * 32];
    x = fmaxf(x, 0.0f);
    v[j] = x; s += x;
  }
  #pragma unroll
  for (int m = 16; m >= 1; m >>= 1) s += __shfl_xor(s, m, 32);
  float mu = s * (1.0f / kH);
  float q = 0.0f;
  #pragma unroll
  for (int j = 0; j < 4; ++j) { float d = v[j] - mu; q += d * d; }
  #pragma unroll
  for (int m = 16; m >= 1; m >>= 1) q += __shfl_xor(q, m, 32);
  float rstd = rsqrtf(q * (1.0f / kH) + kLnEps);
  #pragma unroll
  for (int j = 0; j < 4; ++j) {
    int fidx = lane + j * 32;
    size_t idx = (size_t)row * kH + fidx;
    float nh = h[idx] + (v[j] - mu) * rstd * gamma[fidx] + beta[fidx];
    h[idx] = nh;
    hbf[idx] = (__bf16)nh;
  }
}

// out[row] = out[row] / max(||out[row]||, 1e-12) * g   (rows of 64; one wave per row)
__global__ void rownorm_scale(float* __restrict__ out, const float* __restrict__ g, int n) {
  int row = blockIdx.x * (blockDim.x >> 5) + (threadIdx.x >> 5);
  if (row >= n) return;
  int lane = threadIdx.x & 31;
  float a = out[(size_t)row * kE + lane];
  float b = out[(size_t)row * kE + lane + 32];
  float q = a * a + b * b;
  #pragma unroll
  for (int m = 16; m >= 1; m >>= 1) q += __shfl_xor(q, m, 32);
  float inv = 1.0f / fmaxf(sqrtf(q), 1e-12f);
  out[(size_t)row * kE + lane]      = a * inv * g[lane];
  out[(size_t)row * kE + lane + 32] = b * inv * g[lane + 32];
}

extern "C" void kernel_launch(void* const* d_in, const int* in_sizes, int n_in,
                              void* d_out, int out_size, void* d_ws, size_t ws_size,
                              hipStream_t stream) {
  (void)in_sizes; (void)n_in; (void)out_size; (void)ws_size;
  const int N_SRC = 100000, N_AGT = 50000, D_SRC = 128, D_AGT = 64, N_EDGE = 600000, L = 2;

  const float* x_source = (const float*)d_in[0];
  const float* x_agent  = (const float*)d_in[1];
  const int*   edge_sa  = (const int*)d_in[2];
  const int*   edge_as  = (const int*)d_in[3];
  const float* Win_s  = (const float*)d_in[4];  const float* bin_s = (const float*)d_in[5];
  const float* Win_a  = (const float*)d_in[6];  const float* bin_a = (const float*)d_in[7];
  const float* Wl_sa  = (const float*)d_in[8];  const float* bl_sa = (const float*)d_in[9];
  const float* Wr_sa  = (const float*)d_in[10];
  const float* Wl_as  = (const float*)d_in[11]; const float* bl_as = (const float*)d_in[12];
  const float* Wr_as  = (const float*)d_in[13];
  const float* ln_g_s = (const float*)d_in[14]; const float* ln_b_s = (const float*)d_in[15];
  const float* ln_g_a = (const float*)d_in[16]; const float* ln_b_a = (const float*)d_in[17];
  const float* Wout   = (const float*)d_in[18]; const float* bout  = (const float*)d_in[19];
  const float* gvec   = (const float*)d_in[20];

  // ---- workspace layout: f32 masters first, then bf16 shadows/weights ----
  float* ws = (float*)d_ws;
  size_t off = 0;
  float* hs    = ws + off; off += (size_t)N_SRC * kH;
  float* ha    = ws + off; off += (size_t)N_AGT * kH;
  float* agg_a = ws + off; off += (size_t)N_AGT * kH;
  float* agg_s = ws + off; off += (size_t)N_SRC * kH;
  float* new_a = ws + off; off += (size_t)N_AGT * kH;
  float* new_s = ws + off; off += (size_t)N_SRC * kH;
  float* cnt_a = ws + off; off += N_AGT;
  float* cnt_s = ws + off; off += N_SRC;

  __bf16* bws = (__bf16*)(ws + off);
  size_t boff = 0;
  __bf16* xs_bf     = bws + boff; boff += (size_t)N_SRC * D_SRC;
  __bf16* xa_bf     = bws + boff; boff += (size_t)N_AGT * D_AGT;
  __bf16* hs_bf     = bws + boff; boff += (size_t)N_SRC * kH;
  __bf16* ha_bf     = bws + boff; boff += (size_t)N_AGT * kH;
  __bf16* mean_a_bf = bws + boff; boff += (size_t)N_AGT * kH;
  __bf16* mean_s_bf = bws + boff; boff += (size_t)N_SRC * kH;
  __bf16* WinsT = bws + boff; boff += kH * kH;     // [128 x 128]
  __bf16* WinaT = bws + boff; boff += kH * D_AGT;  // [128 x 64]
  __bf16* WlsaT = bws + boff; boff += (size_t)L * kH * kH;
  __bf16* WrsaT = bws + boff; boff += (size_t)L * kH * kH;
  __bf16* WlasT = bws + boff; boff += (size_t)L * kH * kH;
  __bf16* WrasT = bws + boff; boff += (size_t)L * kH * kH;
  __bf16* WoutT = bws + boff; boff += kE * kH;     // [64 x 128]

  // ---- one-time converts / weight transposes (tiny) ----
  f32_to_bf16<<<(int)(((long long)N_SRC * D_SRC + 255) / 256), 256, 0, stream>>>(x_source, xs_bf, (long long)N_SRC * D_SRC);
  f32_to_bf16<<<(int)(((long long)N_AGT * D_AGT + 255) / 256), 256, 0, stream>>>(x_agent, xa_bf, (long long)N_AGT * D_AGT);
  transpose_w_bf16<<<(kH * kH + 255) / 256, 256, 0, stream>>>(Win_s, WinsT, D_SRC, kH);
  transpose_w_bf16<<<(D_AGT * kH + 255) / 256, 256, 0, stream>>>(Win_a, WinaT, D_AGT, kH);
  for (int i = 0; i < L; ++i) {
    transpose_w_bf16<<<(kH * kH + 255) / 256, 256, 0, stream>>>(Wl_sa + (size_t)i * kH * kH, WlsaT + (size_t)i * kH * kH, kH, kH);
    transpose_w_bf16<<<(kH * kH + 255) / 256, 256, 0, stream>>>(Wr_sa + (size_t)i * kH * kH, WrsaT + (size_t)i * kH * kH, kH, kH);
    transpose_w_bf16<<<(kH * kH + 255) / 256, 256, 0, stream>>>(Wl_as + (size_t)i * kH * kH, WlasT + (size_t)i * kH * kH, kH, kH);
    transpose_w_bf16<<<(kH * kH + 255) / 256, 256, 0, stream>>>(Wr_as + (size_t)i * kH * kH, WrasT + (size_t)i * kH * kH, kH, kH);
  }
  transpose_w_bf16<<<(kH * kE + 255) / 256, 256, 0, stream>>>(Wout, WoutT, kH, kE);

  const int gS = (N_SRC + TILE_M - 1) / TILE_M;   // 1563
  const int gA = (N_AGT + TILE_M - 1) / TILE_M;   // 782

  // Input projections (emit f32 masters + bf16 shadows)
  gemm_wmma_bf16<<<gS, 256, 0, stream>>>(xs_bf, WinsT, nullptr, nullptr,
                                         bin_s, hs, hs_bf, N_SRC, kH, D_SRC, 0);
  gemm_wmma_bf16<<<gA, 256, 0, stream>>>(xa_bf, WinaT, nullptr, nullptr,
                                         bin_a, ha, ha_bf, N_AGT, kH, D_AGT, 0);

  const long long tE = (long long)N_EDGE * kH;
  for (int i = 0; i < L; ++i) {
    hipMemsetAsync(agg_a, 0, (size_t)N_AGT * kH * sizeof(float), stream);
    hipMemsetAsync(agg_s, 0, (size_t)N_SRC * kH * sizeof(float), stream);
    hipMemsetAsync(cnt_a, 0, (size_t)N_AGT * sizeof(float), stream);
    hipMemsetAsync(cnt_s, 0, (size_t)N_SRC * sizeof(float), stream);

    edge_aggregate<<<(int)((tE + 255) / 256), 256, 0, stream>>>(hs, edge_sa, agg_a, cnt_a, N_EDGE);
    edge_aggregate<<<(int)((tE + 255) / 256), 256, 0, stream>>>(ha, edge_as, agg_s, cnt_s, N_EDGE);
    mean_div<<<(int)(((long long)N_AGT * kH + 255) / 256), 256, 0, stream>>>(agg_a, cnt_a, mean_a_bf, N_AGT);
    mean_div<<<(int)(((long long)N_SRC * kH + 255) / 256), 256, 0, stream>>>(agg_s, cnt_s, mean_s_bf, N_SRC);

    // new = mean @ Wl + h_dst @ Wr + bl   (fused dual-GEMM, all-bf16 operands)
    gemm_wmma_bf16<<<gA, 256, 0, stream>>>(mean_a_bf, WlsaT + (size_t)i * kH * kH,
                                           ha_bf,     WrsaT + (size_t)i * kH * kH,
                                           bl_sa + (size_t)i * kH, new_a, nullptr,
                                           N_AGT, kH, kH, kH);
    gemm_wmma_bf16<<<gS, 256, 0, stream>>>(mean_s_bf, WlasT + (size_t)i * kH * kH,
                                           hs_bf,     WrasT + (size_t)i * kH * kH,
                                           bl_as + (size_t)i * kH, new_s, nullptr,
                                           N_SRC, kH, kH, kH);

    relu_ln_res<<<(N_SRC + 7) / 8, 256, 0, stream>>>(new_s, hs, hs_bf, ln_g_s, ln_b_s, N_SRC);
    relu_ln_res<<<(N_AGT + 7) / 8, 256, 0, stream>>>(new_a, ha, ha_bf, ln_g_a, ln_b_a, N_AGT);
  }

  float* out_s = (float*)d_out;
  float* out_a = out_s + (size_t)N_SRC * kE;
  gemm_wmma_bf16<<<gS, 128, 0, stream>>>(hs_bf, WoutT, nullptr, nullptr, bout,
                                         out_s, nullptr, N_SRC, kE, kH, 0);
  gemm_wmma_bf16<<<gA, 128, 0, stream>>>(ha_bf, WoutT, nullptr, nullptr, bout,
                                         out_a, nullptr, N_AGT, kE, kH, 0);
  rownorm_scale<<<(N_SRC + 7) / 8, 256, 0, stream>>>(out_s, gvec, N_SRC);
  rownorm_scale<<<(N_AGT + 7) / 8, 256, 0, stream>>>(out_a, gvec, N_AGT);
}
